// PromptAttention_8907762172395
// MI455X (gfx1250) — matrix operand order repaired and checked by reading, not compiled, not run on hardware
//
#include <hip/hip_runtime.h>

typedef _Float16 half_t;
typedef __attribute__((ext_vector_type(16))) _Float16 v16h;
typedef __attribute__((ext_vector_type(8)))  _Float16 v8h;
typedef __attribute__((ext_vector_type(4)))  _Float16 v4h;
typedef __attribute__((ext_vector_type(8)))  float    v8f;
typedef __attribute__((ext_vector_type(4)))  float    v4f;

#define WMMA_F16(a, b, c) \
  __builtin_amdgcn_wmma_f32_16x16x32_f16(false, (a), false, (b), (short)0, (c), false, false)

#define EMBED   1024
#define HEADS   16
#define HDIM    64
#define BSZ     8
#define TGT     1024
#define SRCL    1024
#define QIN     1024
#define KVIN    256
#define SCALING 0.125f   // 64^-0.5

// ---------------------------------------------------------------------------
// Fragment loaders (wave32; contiguous k order k = 16*khi + s, same in A and B)
// A: 16x32 f16, lane = m + 16*khi; halves[0..7]=A[m][k+khi*16+0..7],
//    halves[8..15]=A[m][k+khi*16+8+0..7]  -> two contiguous 16B loads
// ---------------------------------------------------------------------------
__device__ __forceinline__ v16h load_a_frag(const half_t* __restrict__ base,
                                            int ld, int kbase, int lane) {
  const int ml  = lane & 15;
  const int khi = lane >> 4;
  const half_t* p = base + (size_t)ml * ld + kbase + khi * 16;
  v8h lo = *(const v8h*)(p);
  v8h hi = *(const v8h*)(p + 8);
  return __builtin_shufflevector(lo, hi, 0,1,2,3,4,5,6,7,8,9,10,11,12,13,14,15);
}

// B: 32x16 f16 from B^T[N][K]; lane = n + 16*g; halves[j] = B[g*16+j+kbase][n]
//    = B^T[n][kbase+g*16+j] -> one contiguous 32B load
__device__ __forceinline__ v16h load_b_frag(const half_t* __restrict__ baseT,
                                            int ld, int n0, int kbase, int lane) {
  const int nl = lane & 15;
  const int g  = lane >> 4;
  return *(const v16h*)(baseT + (size_t)(n0 + nl) * ld + kbase + g * 16);
}

// ---------------------------------------------------------------------------
// fp32 -> f16 convert / transpose-convert (weights to N-major)
// ---------------------------------------------------------------------------
__global__ void cvt_f16_kernel(const float* __restrict__ src,
                               half_t* __restrict__ dst, int n) {
  int i = blockIdx.x * blockDim.x + threadIdx.x;
  if (i < n) dst[i] = (half_t)src[i];
}

__global__ void transpose_cvt_kernel(const float* __restrict__ src,  // [K][N]
                                     half_t* __restrict__ dst,       // [N][K]
                                     int K, int N) {
  int i = blockIdx.x * blockDim.x + threadIdx.x;
  if (i < K * N) {
    int k = i / N, n = i - k * N;
    dst[(size_t)n * K + k] = (half_t)src[i];
  }
}

// ---------------------------------------------------------------------------
// WMMA GEMM: C[M=8192][N=1024] = A[M][K] @ B^T[N][K]^T, fp32 accumulate.
// Workgroup = 4 waves; each wave owns a 32(M) x 64(N) tile (2x4 accumulators,
// 8 WMMAs per 32-deep k-step; B fragments amortized over 2 A fragments).
// MODE 0: Q epilogue  ((acc+b)*scaling) -> f16 [b][h][t][d]
// MODE 1: K epilogue  (acc+b)           -> f16 [b][h][s][d]
// MODE 2: V epilogue  (acc+b)           -> f16 [b][h][d][s]  (transposed)
// MODE 3: O epilogue  (acc+b)           -> f32 [bt][n]  (final output)
// ---------------------------------------------------------------------------
template <int MODE>
__global__ __launch_bounds__(128)
void gemm16_kernel(const half_t* __restrict__ A, const half_t* __restrict__ BT,
                   const float* __restrict__ bias,
                   half_t* __restrict__ outH, float* __restrict__ outF, int K) {
  const int lane = threadIdx.x & 31;
  const int wave = threadIdx.x >> 5;
  const int n0   = blockIdx.x * 64;
  const int m0   = blockIdx.y * 128 + wave * 32;
  const int ml   = lane & 15;
  const int gh   = lane >> 4;

  v8f acc[2][4] = {};
  const half_t* arow0 = A  + (size_t)(m0 + ml) * K + gh * 16;
  const half_t* arow1 = arow0 + (size_t)16 * K;
  const half_t* brow  = BT + (size_t)(n0 + ml) * K + gh * 16;

  for (int k = 0; k < K; k += 32) {
    v8h lo0 = *(const v8h*)(arow0 + k);
    v8h hi0 = *(const v8h*)(arow0 + k + 8);
    v8h lo1 = *(const v8h*)(arow1 + k);
    v8h hi1 = *(const v8h*)(arow1 + k + 8);
    v16h a0 = __builtin_shufflevector(lo0, hi0, 0,1,2,3,4,5,6,7,8,9,10,11,12,13,14,15);
    v16h a1 = __builtin_shufflevector(lo1, hi1, 0,1,2,3,4,5,6,7,8,9,10,11,12,13,14,15);
#pragma unroll
    for (int t = 0; t < 4; ++t) {
      v16h b = *(const v16h*)(brow + (size_t)(t * 16) * K + k);
      acc[0][t] = WMMA_F16(a0, b, acc[0][t]);
      acc[1][t] = WMMA_F16(a1, b, acc[1][t]);
    }
  }

#pragma unroll
  for (int mi = 0; mi < 2; ++mi) {
#pragma unroll
    for (int t = 0; t < 4; ++t) {
#pragma unroll
      for (int r = 0; r < 8; ++r) {
        const int m = m0 + mi * 16 + r + 8 * gh;
        const int n = n0 + t * 16 + ml;
        float v = acc[mi][t][r] + bias[n];
        if constexpr (MODE == 0) {
          v *= SCALING;
          const int b = m >> 10, tt = m & 1023, h = n >> 6, d = n & 63;
          outH[(((size_t)(b * HEADS + h) << 10) + tt) * HDIM + d] = (half_t)v;
        } else if constexpr (MODE == 1) {
          const int b = m >> 10, s = m & 1023, h = n >> 6, d = n & 63;
          outH[(((size_t)(b * HEADS + h) << 10) + s) * HDIM + d] = (half_t)v;
        } else if constexpr (MODE == 2) {
          const int b = m >> 10, s = m & 1023, h = n >> 6, d = n & 63;
          outH[((size_t)(b * HEADS + h) * HDIM + d) * SRCL + s] = (half_t)v;
        } else {
          outF[(size_t)m * EMBED + n] = v;
        }
      }
    }
  }
}

// ---------------------------------------------------------------------------
// Fused attention core. One wave per (b, h, 16-row t-tile).
// LDS: 16x1024 fp32 scores (64KB) + 16x1024 f16 unnormalized probs (32KB)
//      + 16 per-row output scales. Scores never touch HBM.
// Normalization (1/rowsum) and the head mask are deferred into the P@V
// epilogue (linear in P), eliminating a whole LDS rescale pass.
// ---------------------------------------------------------------------------
__global__ __launch_bounds__(32)
void attn_kernel(const half_t* __restrict__ q,   // [b][h][t][d]
                 const half_t* __restrict__ k,   // [b][h][s][d]
                 const half_t* __restrict__ vT,  // [b][h][d][s]
                 const float*  __restrict__ mask,// [b][1][t][s]
                 const float*  __restrict__ lhm, // [h]
                 half_t* __restrict__ ao) {      // [b*t][embed] f16
  extern __shared__ char smem[];
  float*  sc     = (float*)smem;                     // 16 x 1024 f32
  half_t* pr     = (half_t*)(smem + 16 * 1024 * 4);  // 16 x 1024 f16
  float*  rscale = (float*)(smem + 96 * 1024);       // 16 f32

  const int lane = threadIdx.x;
  const int bh   = blockIdx.y;          // 0..127
  const int b    = bh >> 4;
  const int h    = bh & 15;
  const int t0   = blockIdx.x * 16;
  const int ml   = lane & 15;
  const int gh   = lane >> 4;

  const half_t* qb = q  + ((size_t)bh * TGT + t0) * HDIM;
  const half_t* kb = k  + (size_t)bh * SRCL * HDIM;
  const half_t* vb = vT + (size_t)bh * HDIM * SRCL;
  const float*  mb = mask + ((size_t)b * TGT + t0) * SRCL;

  // Q fragments (reused for all 64 s-tiles)
  v16h qa0 = load_a_frag(qb, HDIM, 0,  lane);
  v16h qa1 = load_a_frag(qb, HDIM, 32, lane);

  // ---- Phase 1: scores S[16][1024] = Q @ K^T -> LDS ----
  for (int s0 = 0; s0 < SRCL; s0 += 16) {
    v16h kb0 = load_b_frag(kb, HDIM, s0, 0,  lane);
    v16h kb1 = load_b_frag(kb, HDIM, s0, 32, lane);
    v8f c = {};
    c = WMMA_F16(qa0, kb0, c);
    c = WMMA_F16(qa1, kb1, c);
#pragma unroll
    for (int r = 0; r < 8; ++r)
      sc[(size_t)(r + 8 * gh) * SRCL + s0 + ml] = c[r];
  }
  __syncthreads();  // single-wave WG: lowers to waitcnt ordering only

  // ---- Phase 2: softmax. Lane handles row=lane&15, column half gh.
  //      Vectorized b128 accesses, bank-rotated by lane. ----
  {
    const int row = ml;
    const float*  srow = sc + (size_t)row * SRCL + gh * 512;
    const float*  mrow = mb + (size_t)row * SRCL + gh * 512;
    half_t*       prow = pr + (size_t)row * SRCL + gh * 512;

    float mx = -3.0e38f;
    for (int j = 0; j < 128; ++j) {
      const int i = (j + lane) & 127;
      v4f s = *(const v4f*)(srow + 4 * i);
      v4f m = *(const v4f*)(mrow + 4 * i);
      v4f t = s + m;
      mx = fmaxf(mx, fmaxf(fmaxf(t[0], t[1]), fmaxf(t[2], t[3])));
    }
    mx = fmaxf(mx, __shfl_xor(mx, 16, 32));   // combine the two column halves

    float sum = 0.0f;
    for (int j = 0; j < 128; ++j) {
      const int i = (j + lane) & 127;
      v4f s = *(const v4f*)(srow + 4 * i);
      v4f m = *(const v4f*)(mrow + 4 * i);
      float e0 = __expf(s[0] + m[0] - mx);
      float e1 = __expf(s[1] + m[1] - mx);
      float e2 = __expf(s[2] + m[2] - mx);
      float e3 = __expf(s[3] + m[3] - mx);
      sum += (e0 + e1) + (e2 + e3);
      v4h e = { (half_t)e0, (half_t)e1, (half_t)e2, (half_t)e3 };
      *(v4h*)(prow + 4 * i) = e;              // unnormalized (all <= 1.0)
    }
    sum += __shfl_xor(sum, 16, 32);
    if (gh == 0) rscale[row] = lhm[h] / sum;  // head mask folded into 1/sum
  }
  __syncthreads();

  // ---- Phase 3: O[16][64] = scale * (E @ V), write f16 [b*t][h*64+d] ----
  float rs[8];
#pragma unroll
  for (int r = 0; r < 8; ++r) rs[r] = rscale[r + 8 * gh];

  half_t* aob = ao + ((size_t)(b * TGT + t0)) * EMBED + h * HDIM;
#pragma unroll
  for (int t = 0; t < 4; ++t) {
    v8f c = {};
    for (int s = 0; s < SRCL; s += 32) {
      v16h a  = load_a_frag(pr, SRCL, s, lane);            // from LDS
      v16h bf = load_b_frag(vb, SRCL, t * 16, s, lane);    // contiguous vT
      c = WMMA_F16(a, bf, c);
    }
#pragma unroll
    for (int r = 0; r < 8; ++r)
      aob[(size_t)(r + 8 * gh) * EMBED + t * 16 + ml] = (half_t)(c[r] * rs[r]);
  }
}

// ---------------------------------------------------------------------------
extern "C" void kernel_launch(void* const* d_in, const int* in_sizes, int n_in,
                              void* d_out, int out_size, void* d_ws, size_t ws_size,
                              hipStream_t stream) {
  const float* hs   = (const float*)d_in[0];   // [8,1024,1024]
  const float* kv   = (const float*)d_in[1];   // [8,1024,256]
  const float* mask = (const float*)d_in[2];   // [8,1,1024,1024]
  const float* lhm  = (const float*)d_in[3];   // [16]
  const float* Wq   = (const float*)d_in[4];   // [1024,1024]
  const float* bq   = (const float*)d_in[5];
  const float* Wk   = (const float*)d_in[6];   // [256,1024]
  const float* bk   = (const float*)d_in[7];
  const float* Wv   = (const float*)d_in[8];   // [256,1024]
  const float* bv   = (const float*)d_in[9];
  const float* Wo   = (const float*)d_in[10];  // [1024,1024]
  const float* bo   = (const float*)d_in[11];
  float* out = (float*)d_out;

  char* ws = (char*)d_ws;
  size_t off = 0;
  auto alloc = [&](size_t bytes) -> char* {
    char* p = ws + off;
    off += (bytes + 255) & ~(size_t)255;
    return p;
  };

  half_t* hsf  = (half_t*)alloc((size_t)BSZ * TGT  * QIN  * 2);  // 16 MB
  half_t* kvf  = (half_t*)alloc((size_t)BSZ * SRCL * KVIN * 2);  //  4 MB
  half_t* WqT  = (half_t*)alloc((size_t)EMBED * QIN  * 2);       //  2 MB
  half_t* WkT  = (half_t*)alloc((size_t)EMBED * KVIN * 2);       // .5 MB
  half_t* WvT  = (half_t*)alloc((size_t)EMBED * KVIN * 2);       // .5 MB
  half_t* WoT  = (half_t*)alloc((size_t)EMBED * EMBED * 2);      //  2 MB
  half_t* qbuf = (half_t*)alloc((size_t)BSZ * HEADS * TGT  * HDIM * 2); // 16 MB
  half_t* kbuf = (half_t*)alloc((size_t)BSZ * HEADS * SRCL * HDIM * 2); // 16 MB
  half_t* vTbuf= (half_t*)alloc((size_t)BSZ * HEADS * HDIM * SRCL * 2); // 16 MB
  half_t* aobuf= (half_t*)alloc((size_t)BSZ * TGT * EMBED * 2);         // 16 MB
  (void)ws_size; (void)in_sizes; (void)n_in; (void)out_size;

  // 1) conversions / weight transposes
  {
    int n = BSZ * TGT * QIN;
    cvt_f16_kernel<<<(n + 255) / 256, 256, 0, stream>>>(hs, hsf, n);
  }
  {
    int n = BSZ * SRCL * KVIN;
    cvt_f16_kernel<<<(n + 255) / 256, 256, 0, stream>>>(kv, kvf, n);
  }
  transpose_cvt_kernel<<<(QIN * EMBED + 255) / 256, 256, 0, stream>>>(Wq, WqT, QIN,  EMBED);
  transpose_cvt_kernel<<<(KVIN * EMBED + 255) / 256, 256, 0, stream>>>(Wk, WkT, KVIN, EMBED);
  transpose_cvt_kernel<<<(KVIN * EMBED + 255) / 256, 256, 0, stream>>>(Wv, WvT, KVIN, EMBED);
  transpose_cvt_kernel<<<(EMBED * EMBED + 255) / 256, 256, 0, stream>>>(Wo, WoT, EMBED, EMBED);

  // 2) projections (M=8192, N=1024; 128x64 workgroup tiles, 32x64 per wave)
  dim3 ggrid(EMBED / 64, (BSZ * TGT) / 128);
  gemm16_kernel<0><<<ggrid, 128, 0, stream>>>(hsf, WqT, bq, qbuf,  nullptr, QIN);
  gemm16_kernel<1><<<ggrid, 128, 0, stream>>>(kvf, WkT, bk, kbuf,  nullptr, KVIN);
  gemm16_kernel<2><<<ggrid, 128, 0, stream>>>(kvf, WvT, bv, vTbuf, nullptr, KVIN);

  // 3) fused attention: one wave per (b, h, 16 t-rows); ~96 KB LDS per WG
  attn_kernel<<<dim3(TGT / 16, BSZ * HEADS), 32, 96 * 1024 + 256, stream>>>(
      qbuf, kbuf, vTbuf, mask, lhm, aobuf);

  // 4) output projection -> fp32 d_out
  gemm16_kernel<3><<<ggrid, 128, 0, stream>>>(aobuf, WoT, bo, nullptr, out, EMBED);
}
